// Transformer_42984032698615
// MI455X (gfx1250) — compile-verified, matrix operand order checked by reading, not run on hardware
//
#include <hip/hip_runtime.h>

// ---------------- types ----------------
typedef __bf16 bf16_t;
typedef __attribute__((ext_vector_type(16))) __bf16 v16bf;
typedef __attribute__((ext_vector_type(8)))  __bf16 v8bf;
typedef __attribute__((ext_vector_type(8)))  float  v8f;
typedef __attribute__((ext_vector_type(4)))  unsigned int u32x4;
typedef __attribute__((ext_vector_type(8)))  int i32x8;
typedef __attribute__((ext_vector_type(4)))  int i32x4;

// TDM availability (device pass only)
#if defined(__AMDGCN__) && defined(__has_builtin)
#  if __has_builtin(__builtin_amdgcn_tensor_load_to_lds) && \
      __has_builtin(__builtin_amdgcn_s_wait_tensorcnt)
#    define HAVE_TDM 1
#  endif
#endif
#ifndef HAVE_TDM
#  define HAVE_TDM 0
#endif

// ---------------- constants ----------------
constexpr int C_B  = 2;
constexpr int C_S  = 2048;
constexpr int C_D  = 512;
constexpr int C_F  = 2048;
constexpr int C_H  = 8;
constexpr int C_DH = 64;
constexpr int C_V  = 32000;
constexpr int C_L  = 4;
constexpr int C_N  = C_B * C_S;   // 4096 token rows

__device__ __forceinline__ bf16_t f2bf(float f) {
    unsigned u = __float_as_uint(f);
    u += 0x7FFFu + ((u >> 16) & 1u);          // round-to-nearest-even
    unsigned short h = (unsigned short)(u >> 16);
    return __builtin_bit_cast(bf16_t, h);
}

union FragU { v16bf v; v8bf h2[2]; };

#if HAVE_TDM
// Issue a 2D tensor DMA: global (row-major, row stride = stride elems, 2B elems)
// -> LDS packed tile [tile_h][tile_w]. Descriptor per CDNA5 ISA 8.3/8.4.
__device__ __forceinline__ void tdm_load_2d(unsigned lds_addr, const void* gptr,
                                            unsigned tile_w, unsigned tile_h,
                                            unsigned stride_elems) {
    unsigned long long ga = (unsigned long long)(uintptr_t)gptr;
    u32x4 g0;
    g0[0] = 1u;                                            // count=1, load, no gather
    g0[1] = lds_addr;                                      // LDS byte address
    g0[2] = (unsigned)(ga & 0xFFFFFFFFu);                  // global_addr[31:0]
    g0[3] = (unsigned)((ga >> 32) & 0x01FFFFFFu) | 0x80000000u; // addr[56:32] | type=2
    unsigned long long tw = stride_elems;                  // tensor_dim0 = full row
    unsigned long long th = tile_h;                        // tensor_dim1
    unsigned long long u0 = ((unsigned long long)(1u << 16))        // data_size=1 (2B)
                          | ((tw & 0xFFFFull) << 48);               // tensor_dim0[15:0]
    unsigned long long u1 = ((tw >> 16) & 0xFFFFull)                // tensor_dim0[31:16]
                          | ((th & 0xFFFFFFFFull) << 16)            // tensor_dim1
                          | (((unsigned long long)tile_w & 0xFFFFull) << 48); // tile_dim0
    unsigned long long u2 = ((unsigned long long)tile_h & 0xFFFFull)          // tile_dim1
                          | (((unsigned long long)stride_elems & 0xFFFFFFFFull) << 32);
    unsigned long long u3 = ((unsigned long long)stride_elems >> 32) & 0xFFFFull;
    i32x8 g1;
    g1[0] = (int)(u0 & 0xFFFFFFFFull); g1[1] = (int)(u0 >> 32);
    g1[2] = (int)(u1 & 0xFFFFFFFFull); g1[3] = (int)(u1 >> 32);
    g1[4] = (int)(u2 & 0xFFFFFFFFull); g1[5] = (int)(u2 >> 32);
    g1[6] = (int)(u3 & 0xFFFFFFFFull); g1[7] = (int)(u3 >> 32);
    i32x4 gz = {0, 0, 0, 0};
#if __clang_major__ >= 23
    i32x8 gz8 = {0, 0, 0, 0, 0, 0, 0, 0};
    __builtin_amdgcn_tensor_load_to_lds(g0, g1, gz, gz, gz8, 0);
#else
    __builtin_amdgcn_tensor_load_to_lds(g0, g1, gz, gz, 0);
#endif
}
#endif

// ---------------- elementwise kernels ----------------
__global__ void cvt_kernel(const float* __restrict__ src, bf16_t* __restrict__ dst, int n) {
    int i = blockIdx.x * 256 + threadIdx.x;
    if (i < n) dst[i] = f2bf(src[i]);
}

// Transposing weight conversion: dst[n*K + k] = bf16(W(k,n)).
// BMODE=1: source is head-blocked (h, d, e) with n = h*64+e, k = d.
template<int BMODE>
__global__ void cvtT_kernel(const float* __restrict__ src, bf16_t* __restrict__ dst,
                            int N, int K) {
    long long i = (long long)blockIdx.x * 256 + threadIdx.x;
    if (i >= (long long)N * K) return;
    int n = (int)(i / K), k = (int)(i % K);
    size_t soff = BMODE ? ((size_t)(n >> 6)) * ((size_t)K * C_DH) + (size_t)k * C_DH + (n & 63)
                        : (size_t)k * N + n;
    dst[i] = f2bf(src[soff]);
}

__global__ void embed_kernel(const int* __restrict__ X, const float* __restrict__ E,
                             float* __restrict__ x) {
    int row = blockIdx.x, t = threadIdx.x;
    int tok = X[row];
    const float* er = E + (size_t)tok * C_D;
    float* xr = x + (size_t)row * C_D;
    xr[t]       = er[t];
    xr[t + 256] = er[t + 256];
}

__global__ void ln_kernel(const float* __restrict__ x, const float* __restrict__ g,
                          const float* __restrict__ b, bf16_t* __restrict__ out) {
    __shared__ float red[256];
    int row = blockIdx.x, t = threadIdx.x;
    const float* xr = x + (size_t)row * C_D;
    float v0 = xr[t], v1 = xr[t + 256];
    red[t] = v0 + v1; __syncthreads();
    for (int s = 128; s > 0; s >>= 1) { if (t < s) red[t] += red[t + s]; __syncthreads(); }
    float mean = red[0] * (1.0f / C_D);
    __syncthreads();
    float d0 = v0 - mean, d1 = v1 - mean;
    red[t] = d0 * d0 + d1 * d1; __syncthreads();
    for (int s = 128; s > 0; s >>= 1) { if (t < s) red[t] += red[t + s]; __syncthreads(); }
    float inv = rsqrtf(red[0] * (1.0f / C_D) + 1e-5f);
    bf16_t* orow = out + (size_t)row * C_D;
    orow[t]       = f2bf(d0 * inv * g[t]       + b[t]);
    orow[t + 256] = f2bf(d1 * inv * g[t + 256] + b[t + 256]);
}

// ---------------- bf16 WMMA GEMM ----------------
// C[M,N] = A[M,K] * Wt[N,K]^T + bias (Wt pre-transposed [N][K]).
// Macro tile 64x64, K-step 32; 8 waves in 4(M) x 2(N), each wave computes
// one 16(M) x 32(N) strip = 2 WMMA tiles sharing the A fragment.
// TDM path: double-buffered LDS tiles; wave 0 issues next step's two
// tensor_load_to_lds then waits TENSORcnt<=2 (in-order decrement) so the
// current buffer's DMA pair has landed while the next overlaps compute.
// OMODE: 0 = row-major [M][N], 1 = scatter (B,H,S,DH), 2 = scatter (B,H,DH,S).
template<int OMODE, int ACT, bool RESID, bool OUTF, bool OUTB>
__global__ void gemm_kernel(const bf16_t* __restrict__ A, const bf16_t* __restrict__ Wt,
                            const float* __restrict__ bias, const float* __restrict__ resid,
                            float* __restrict__ outf, bf16_t* __restrict__ outb,
                            int M, int N, int K) {
    __shared__ bf16_t As[2][64 * 32];   // [m][k]
    __shared__ bf16_t Bs[2][64 * 32];   // [n][k]
    int t = threadIdx.x, wave = t >> 5, lane = t & 31, lr = lane & 15, hi = lane >> 4;
    int bm = blockIdx.y * 64, bn = blockIdx.x * 64;
    int wm = wave >> 1, wn = wave & 1;
    v8f acc0 = {0.f, 0.f, 0.f, 0.f, 0.f, 0.f, 0.f, 0.f};
    v8f acc1 = {0.f, 0.f, 0.f, 0.f, 0.f, 0.f, 0.f, 0.f};
    const bf16_t* gA = A  + (size_t)bm * K;
    const bf16_t* gB = Wt + (size_t)bn * K;
#if HAVE_TDM
    unsigned ldsA[2] = { (unsigned)(uintptr_t)(void*)&As[0][0],
                         (unsigned)(uintptr_t)(void*)&As[1][0] };
    unsigned ldsB[2] = { (unsigned)(uintptr_t)(void*)&Bs[0][0],
                         (unsigned)(uintptr_t)(void*)&Bs[1][0] };
    if (wave == 0) {                     // prologue: fill buffer 0
        tdm_load_2d(ldsA[0], gA, 32, 64, (unsigned)K);
        tdm_load_2d(ldsB[0], gB, 32, 64, (unsigned)K);
    }
#else
    int row = t >> 2, seg = t & 3;       // one v8bf per thread per tile
#endif
    for (int k0 = 0; k0 < K; k0 += 32) {
#if HAVE_TDM
        int buf = (k0 >> 5) & 1;
        if (wave == 0) {
            if (k0 + 32 < K) {           // prefetch next step into other buffer
                tdm_load_2d(ldsA[buf ^ 1], gA + k0 + 32, 32, 64, (unsigned)K);
                tdm_load_2d(ldsB[buf ^ 1], gB + k0 + 32, 32, 64, (unsigned)K);
                __builtin_amdgcn_s_wait_tensorcnt(2);
            } else {
                __builtin_amdgcn_s_wait_tensorcnt(0);
            }
        }
#else
        const int buf = 0;
        ((v8bf*)As[0])[t] = *(const v8bf*)(gA + (size_t)row * K + k0 + seg * 8);
        ((v8bf*)Bs[0])[t] = *(const v8bf*)(gB + (size_t)row * K + k0 + seg * 8);
        if (k0 + 32 < K) {
            __builtin_prefetch(gA + (size_t)row * K + k0 + 32, 0, 1);
            __builtin_prefetch(gB + (size_t)row * K + k0 + 32, 0, 1);
        }
#endif
        __syncthreads();
        FragU ua, ub0, ub1;
        const v8bf* pa = (const v8bf*)(As[buf] + (wm * 16 + lr) * 32);
        ua.h2[0]  = pa[hi];  ua.h2[1]  = pa[2 + hi];
        const v8bf* pb0 = (const v8bf*)(Bs[buf] + (wn * 32 + lr) * 32);
        ub0.h2[0] = pb0[hi]; ub0.h2[1] = pb0[2 + hi];
        const v8bf* pb1 = (const v8bf*)(Bs[buf] + (wn * 32 + 16 + lr) * 32);
        ub1.h2[0] = pb1[hi]; ub1.h2[1] = pb1[2 + hi];
        acc0 = __builtin_amdgcn_wmma_f32_16x16x32_bf16(false, ua.v, false, ub0.v,
                                                       (short)0, acc0, false, false);
        acc1 = __builtin_amdgcn_wmma_f32_16x16x32_bf16(false, ua.v, false, ub1.v,
                                                       (short)0, acc1, false, false);
        __syncthreads();
    }
    // epilogue
    int gn0 = bn + wn * 32 + lr;
    float bv0 = bias[gn0], bv1 = bias[gn0 + 16];
    #pragma unroll
    for (int r = 0; r < 8; r++) {
        int gm = bm + wm * 16 + r + (hi << 3);
        float v0 = acc0[r] + bv0;
        float v1 = acc1[r] + bv1;
        if constexpr (RESID) {
            v0 += resid[(size_t)gm * N + gn0];
            v1 += resid[(size_t)gm * N + gn0 + 16];
        }
        if constexpr (ACT) {  // exact GELU
            v0 = 0.5f * v0 * (1.0f + erff(v0 * 0.70710678118f));
            v1 = 0.5f * v1 * (1.0f + erff(v1 * 0.70710678118f));
        }
        if constexpr (OUTF) {
            outf[(size_t)gm * N + gn0]      = v0;
            outf[(size_t)gm * N + gn0 + 16] = v1;
        }
        if constexpr (OUTB) {
            int b2 = gm >> 11, s = gm & (C_S - 1);
            int h0 = gn0 >> 6, e0 = gn0 & 63;
            int h1 = (gn0 + 16) >> 6, e1 = (gn0 + 16) & 63;
            if constexpr (OMODE == 1) {        // (B,H,S,DH)
                outb[(((size_t)(b2 * C_H + h0)) * C_S + s) * C_DH + e0] = f2bf(v0);
                outb[(((size_t)(b2 * C_H + h1)) * C_S + s) * C_DH + e1] = f2bf(v1);
            } else if constexpr (OMODE == 2) { // (B,H,DH,S)  -- V^T for attention
                outb[(((size_t)(b2 * C_H + h0)) * C_DH + e0) * C_S + s] = f2bf(v0);
                outb[(((size_t)(b2 * C_H + h1)) * C_DH + e1) * C_S + s] = f2bf(v1);
            } else {
                outb[(size_t)gm * N + gn0]      = f2bf(v0);
                outb[(size_t)gm * N + gn0 + 16] = f2bf(v1);
            }
        }
    }
}

// ---------------- fused causal attention ----------------
// Grid (S/16, H, B), 256 threads (8 waves). Whole 16xS score stripe in LDS.
// Vt is pre-transposed (B,H,DH,S) so AV B-fragments load straight from global.
__global__ void attn_kernel(const bf16_t* __restrict__ Q, const bf16_t* __restrict__ Kb,
                            const bf16_t* __restrict__ Vt, bf16_t* __restrict__ O) {
    extern __shared__ char smem[];
    float*  sc   = (float*)smem;                                      // 16 x 2048 f32
    bf16_t* pb   = (bf16_t*)(smem + 16 * 2048 * 4);                   // 16 x 2048 bf16
    float*  red  = (float*)(smem + 16 * 2048 * 4 + 16 * 2048 * 2);    // 16 x 16
    float*  redc = (float*)(smem + 16 * 2048 * 4 + 16 * 2048 * 2 + 16 * 16 * 4); // 4x32x8

    int qt = blockIdx.x, h = blockIdx.y, b = blockIdx.z;
    int t = threadIdx.x, wave = t >> 5, lane = t & 31, lr = lane & 15, hi = lane >> 4;
    size_t headbase = ((size_t)(b * C_H + h)) * C_S * C_DH;
    int q0 = qt * 16;

    const v8bf* qp = (const v8bf*)(Q + headbase + (size_t)(q0 + lr) * C_DH);
    FragU aq0, aq1;
    aq0.h2[0] = qp[hi];     aq0.h2[1] = qp[2 + hi];
    aq1.h2[0] = qp[4 + hi]; aq1.h2[1] = qp[6 + hi];

    const float scale = 0.125f;  // 1/sqrt(64)
    for (int kt = wave; kt < C_S / 16; kt += 8) {
        const v8bf* kp = (const v8bf*)(Kb + headbase + (size_t)(kt * 16 + lr) * C_DH);
        FragU bk0, bk1;
        bk0.h2[0] = kp[hi];     bk0.h2[1] = kp[2 + hi];
        bk1.h2[0] = kp[4 + hi]; bk1.h2[1] = kp[6 + hi];
        v8f c = {0.f, 0.f, 0.f, 0.f, 0.f, 0.f, 0.f, 0.f};
        c = __builtin_amdgcn_wmma_f32_16x16x32_bf16(false, aq0.v, false, bk0.v,
                                                    (short)0, c, false, false);
        c = __builtin_amdgcn_wmma_f32_16x16x32_bf16(false, aq1.v, false, bk1.v,
                                                    (short)0, c, false, false);
        int kg = kt * 16 + lr;
        #pragma unroll
        for (int r = 0; r < 8; r++) {
            int m = r + (hi << 3);
            float sv = c[r] * scale;
            if (kg > q0 + m) sv = -1000000000.0f;   // causal mask
            sc[m * C_S + kg] = sv;
        }
    }
    __syncthreads();

    // softmax: 16 threads per row
    int row = t >> 4, j0 = t & 15;
    float mx = -3.0e38f;
    for (int j = j0; j < C_S; j += 16) mx = fmaxf(mx, sc[row * C_S + j]);
    red[row * 16 + j0] = mx; __syncthreads();
    float rmax = -3.0e38f;
    #pragma unroll
    for (int i = 0; i < 16; i++) rmax = fmaxf(rmax, red[row * 16 + i]);
    __syncthreads();
    float ssum = 0.0f;
    for (int j = j0; j < C_S; j += 16) {
        float e = __expf(sc[row * C_S + j] - rmax);
        sc[row * C_S + j] = e; ssum += e;
    }
    red[row * 16 + j0] = ssum; __syncthreads();
    float rs = 0.0f;
    #pragma unroll
    for (int i = 0; i < 16; i++) rs += red[row * 16 + i];
    float inv = 1.0f / rs;
    for (int j = j0; j < C_S; j += 16) pb[row * C_S + j] = f2bf(sc[row * C_S + j] * inv);
    __syncthreads();

    // P*V : 8 waves = 4 DH-column tiles x 2 key halves; B-frags from global Vt.
    v8f acc = {0.f, 0.f, 0.f, 0.f, 0.f, 0.f, 0.f, 0.f};
    int n0 = (wave & 3) * 16;
    int kh = wave >> 2;                      // key half: 0 or 1
    const v8bf* vrow = (const v8bf*)(Vt + (((size_t)(b * C_H + h)) * C_DH + n0 + lr) * C_S
                                        + kh * 1024);
    const v8bf* prow = (const v8bf*)(pb + lr * C_S + kh * 1024);
    for (int kk = 0; kk < 32; kk++) {        // 32 steps x 32 keys
        FragU ap, bvv;
        ap.h2[0]  = prow[kk * 4 + hi];  ap.h2[1]  = prow[kk * 4 + 2 + hi];
        bvv.h2[0] = vrow[kk * 4 + hi];  bvv.h2[1] = vrow[kk * 4 + 2 + hi];
        acc = __builtin_amdgcn_wmma_f32_16x16x32_bf16(false, ap.v, false, bvv.v,
                                                      (short)0, acc, false, false);
    }
    // cross-half reduction via LDS, then store concat-head layout (B,S,D)
    if (wave >= 4) {
        #pragma unroll
        for (int r = 0; r < 8; r++) redc[((wave - 4) * 32 + lane) * 8 + r] = acc[r];
    }
    __syncthreads();
    if (wave < 4) {
        #pragma unroll
        for (int r = 0; r < 8; r++) {
            int m = r + (hi << 3);
            float v = acc[r] + redc[(wave * 32 + lane) * 8 + r];
            O[((size_t)(b * C_S + q0 + m)) * C_D + h * C_DH + n0 + lr] = f2bf(v);
        }
    }
}

// ---------------- host orchestration ----------------
extern "C" void kernel_launch(void* const* d_in, const int* in_sizes, int n_in,
                              void* d_out, int out_size, void* d_ws, size_t ws_size,
                              hipStream_t stream) {
    (void)in_sizes; (void)n_in; (void)out_size; (void)ws_size;
    const int*   X     = (const int*)d_in[0];
    const float* embed = (const float*)d_in[1];
    const float* Wq_f  = (const float*)d_in[2];
    const float* bq_f  = (const float*)d_in[3];
    const float* Wk_f  = (const float*)d_in[4];
    const float* bk_f  = (const float*)d_in[5];
    const float* Wv_f  = (const float*)d_in[6];
    const float* bv_f  = (const float*)d_in[7];
    const float* Wo_f  = (const float*)d_in[8];
    const float* bo_f  = (const float*)d_in[9];
    const float* ln1g  = (const float*)d_in[10];
    const float* ln1b  = (const float*)d_in[11];
    const float* ln2g  = (const float*)d_in[12];
    const float* ln2b  = (const float*)d_in[13];
    const float* W1_f  = (const float*)d_in[14];
    const float* b1_f  = (const float*)d_in[15];
    const float* W2_f  = (const float*)d_in[16];
    const float* b2_f  = (const float*)d_in[17];
    const float* Wp_f  = (const float*)d_in[18];
    const float* bp_f  = (const float*)d_in[19];
    float* out = (float*)d_out;

    char* ws = (char*)d_ws;
    size_t off = 0;
    auto take = [&](size_t bytes) -> char* {
        char* p = ws + off; off += (bytes + 255) & ~(size_t)255; return p;
    };
    const size_t nDD = (size_t)C_L * C_D * C_D;
    const size_t nDF = (size_t)C_L * C_D * C_F;
    bf16_t* wq  = (bf16_t*)take(nDD * 2);            // transposed [N][K] per layer
    bf16_t* wk  = (bf16_t*)take(nDD * 2);
    bf16_t* wv  = (bf16_t*)take(nDD * 2);
    bf16_t* wo  = (bf16_t*)take(nDD * 2);
    bf16_t* w1  = (bf16_t*)take(nDF * 2);
    bf16_t* w2  = (bf16_t*)take(nDF * 2);
    bf16_t* wp  = (bf16_t*)take((size_t)C_D * C_V * 2);
    float*  x   = (float*) take((size_t)C_N * C_D * 4);
    float*  y   = (float*) take((size_t)C_N * C_D * 4);
    bf16_t* hbf = (bf16_t*)take((size_t)C_N * C_D * 2);
    bf16_t* qbf = (bf16_t*)take((size_t)C_N * C_D * 2);
    bf16_t* kbf = (bf16_t*)take((size_t)C_N * C_D * 2);
    bf16_t* vbf = (bf16_t*)take((size_t)C_N * C_D * 2);  // (B,H,DH,S)
    bf16_t* obf = (bf16_t*)take((size_t)C_N * C_D * 2);
    bf16_t* tbf = (bf16_t*)take((size_t)C_N * C_F * 2);

    auto cvtT = [&](int bmode, const float* s, bf16_t* d, int N, int K) {
        unsigned blocks = (unsigned)(((long long)N * K + 255) / 256);
        if (bmode) cvtT_kernel<1><<<blocks, 256, 0, stream>>>(s, d, N, K);
        else       cvtT_kernel<0><<<blocks, 256, 0, stream>>>(s, d, N, K);
    };

    // weight conversion + transpose to [N][K] bf16, once per launch
    for (int l = 0; l < C_L; l++) {
        size_t oDD = (size_t)l * C_D * C_D, oDF = (size_t)l * C_D * C_F;
        cvtT(1, Wq_f + oDD, wq + oDD, C_D, C_D);
        cvtT(1, Wk_f + oDD, wk + oDD, C_D, C_D);
        cvtT(1, Wv_f + oDD, wv + oDD, C_D, C_D);
        cvtT(0, Wo_f + oDD, wo + oDD, C_D, C_D);
        cvtT(0, W1_f + oDF, w1 + oDF, C_F, C_D);   // dst [F][D]
        cvtT(0, W2_f + oDF, w2 + oDF, C_D, C_F);   // dst [D][F]
    }
    cvtT(0, Wp_f, wp, C_V, C_D);                   // dst [V][D]

    embed_kernel<<<C_N, 256, 0, stream>>>(X, embed, x);

    const size_t attn_smem = 16 * 2048 * 4 + 16 * 2048 * 2 + 16 * 16 * 4 + 4 * 32 * 8 * 4;
    for (int l = 0; l < C_L; l++) {
        size_t oDD = (size_t)l * C_D * C_D, oDF = (size_t)l * C_D * C_F;
        ln_kernel<<<C_N, 256, 0, stream>>>(x, ln1g + l * C_D, ln1b + l * C_D, hbf);
        gemm_kernel<1, 0, false, false, true><<<dim3(C_D / 64, C_N / 64), 256, 0, stream>>>(
            hbf, wq + oDD, bq_f + l * C_D, nullptr, nullptr, qbf, C_N, C_D, C_D);
        gemm_kernel<1, 0, false, false, true><<<dim3(C_D / 64, C_N / 64), 256, 0, stream>>>(
            hbf, wk + oDD, bk_f + l * C_D, nullptr, nullptr, kbf, C_N, C_D, C_D);
        gemm_kernel<2, 0, false, false, true><<<dim3(C_D / 64, C_N / 64), 256, 0, stream>>>(
            hbf, wv + oDD, bv_f + l * C_D, nullptr, nullptr, vbf, C_N, C_D, C_D);
        attn_kernel<<<dim3(C_S / 16, C_H, C_B), 256, attn_smem, stream>>>(qbf, kbf, vbf, obf);
        gemm_kernel<0, 0, true, true, false><<<dim3(C_D / 64, C_N / 64), 256, 0, stream>>>(
            obf, wo + oDD, bo_f + l * C_D, x, y, nullptr, C_N, C_D, C_D);
        ln_kernel<<<C_N, 256, 0, stream>>>(y, ln2g + l * C_D, ln2b + l * C_D, hbf);
        gemm_kernel<0, 1, false, false, true><<<dim3(C_F / 64, C_N / 64), 256, 0, stream>>>(
            hbf, w1 + oDF, b1_f + l * C_F, nullptr, nullptr, tbf, C_N, C_F, C_D);
        gemm_kernel<0, 0, true, true, false><<<dim3(C_D / 64, C_N / 64), 256, 0, stream>>>(
            tbf, w2 + oDF, b2_f + l * C_D, y, x, nullptr, C_N, C_D, C_F);
    }
    // LM head
    cvt_kernel<<<(C_N * C_D + 255) / 256, 256, 0, stream>>>(x, hbf, C_N * C_D);
    gemm_kernel<0, 0, false, true, false><<<dim3(C_V / 64, C_N / 64), 256, 0, stream>>>(
        hbf, wp, bp_f, nullptr, out, nullptr, C_N, C_V, C_D);
}
